// GAT_BN_60859686584881
// MI455X (gfx1250) — matrix operand order repaired and checked by reading, not compiled, not run on hardware
//
#include <hip/hip_runtime.h>

// ---------------- problem constants (match reference) ----------------
#define NNODES 50000
#define NEDGES 800000
#define F_INC  512
#define HIDC   16
#define NHEAD  8
#define HCC    128   // NHEAD * HIDC
#define LBL    40
#define NEG_SLOPE 0.2f
#define ELU_ALPHA 0.2f
#define BN_EPS    1e-5f

typedef __attribute__((ext_vector_type(2)))  float    v2f;
typedef __attribute__((ext_vector_type(8)))  float    v8f;
typedef __attribute__((ext_vector_type(16))) _Float16 v16h;

// =====================================================================
// GEMM: Y[nrows,kout] = X[nrows,kin] @ W[kin,kout]   (f32, WMMA)
// Block = 256 threads = 8 waves. blockIdx.y picks a 16-column panel of W
// which is staged in LDS once and shared; each wave computes one 16x16
// output tile using V_WMMA_F32_16X16X4_F32 (full f32 precision).
// =====================================================================
__global__ void __launch_bounds__(256) gat_gemm_wmma(
    const float* __restrict__ X, const float* __restrict__ W,
    float* __restrict__ Y, int nrows, int kin, int kout) {
  extern __shared__ float ldsW[];  // [kin][16] column panel
  const int col0 = blockIdx.y * 16;
  for (int i = threadIdx.x; i < kin * 16; i += blockDim.x) {
    const int k = i >> 4, c = i & 15;
    const int col = col0 + c;
    ldsW[i] = (col < kout) ? W[(size_t)k * kout + col] : 0.0f;
  }
  __syncthreads();

  const int wave = threadIdx.x >> 5;
  const int lane = threadIdx.x & 31;
  const int half = lane >> 4;   // lane half selects K sub-column
  const int lsub = lane & 15;   // M (for A) / N (for B,C)
  const int row0 = (blockIdx.x * 8 + wave) * 16;
  if (row0 >= nrows) return;    // wave-uniform exit (EXEC stays all-ones)

  const int row  = row0 + lsub;
  const int rowc = row < nrows ? row : nrows - 1;  // clamp, stores are guarded
  const float* __restrict__ xrow = X + (size_t)rowc * kin;

  v8f acc = {};
#if !defined(__HIP_DEVICE_COMPILE__)
  (void)xrow;  // host stub; never executed
#elif __has_builtin(__builtin_amdgcn_wmma_f32_16x16x4_f32)
  // f32 A 16x4 layout: VGPR0 = {K0 | K2}, VGPR1 = {K1 | K3} (per lane-half)
  for (int k0 = 0; k0 < kin; k0 += 4) {
    const int ka = k0 + 2 * half;
    v2f a, b;
    a.x = xrow[ka];
    a.y = xrow[ka + 1];
    b.x = ldsW[ka * 16 + lsub];
    b.y = ldsW[(ka + 1) * 16 + lsub];
    acc = __builtin_amdgcn_wmma_f32_16x16x4_f32(false, a, false, b, (short)0,
                                                acc, false, false);
  }
#else
  // Fallback: codegen-confirmed f16 WMMA with f32 accumulation.
  for (int k0 = 0; k0 < kin; k0 += 32) {
    v16h a, b;
#pragma unroll
    for (int j = 0; j < 16; ++j) {
      const int v = j >> 1, jj = j & 1;
      const int ka = k0 + ((v >> 2) * 16) + (v & 3) * 2 + jj + 8 * half;
      a[j] = (_Float16)xrow[ka];
      const int kb = k0 + ((j >> 3) * 16) + (j & 7) + 8 * half;
      b[j] = (_Float16)ldsW[kb * 16 + lsub];
    }
    acc = __builtin_amdgcn_wmma_f32_16x16x32_f16(false, a, false, b, (short)0,
                                                 acc, false, false);
  }
#endif

  // C/D layout: VGPR r -> M = r + 8*half, N = lsub
#pragma unroll
  for (int r = 0; r < 8; ++r) {
    const int rout = row0 + r + 8 * half;
    const int col  = col0 + lsub;
    if (rout < nrows && col < kout)
      Y[(size_t)rout * kout + col] = acc[r];
  }
}

// =====================================================================
// helpers / elementwise kernels
// =====================================================================
__global__ void fill_f32(float* __restrict__ p, int n, float v) {
  const int t = blockIdx.x * blockDim.x + threadIdx.x;
  if (t < n) p[t] = v;
}

// alpha_src / alpha_dst per (node, head)
__global__ void gat_attn_logits(const float* __restrict__ hbuf,
                                const float* __restrict__ asv,
                                const float* __restrict__ adv,
                                float* __restrict__ als, float* __restrict__ ald,
                                int n, int Hn, int HIDn, int HCn) {
  const int t = blockIdx.x * blockDim.x + threadIdx.x;
  if (t >= n * Hn) return;
  const int node = t / Hn, h = t - node * Hn;
  const float* __restrict__ hp  = hbuf + (size_t)node * HCn + h * HIDn;
  const float* __restrict__ asp = asv + h * HIDn;
  const float* __restrict__ adp = adv + h * HIDn;
  float ss = 0.f, sd = 0.f;
  for (int c = 0; c < HIDn; ++c) {
    const float v = hp[c];
    ss += v * asp[c];
    sd += v * adp[c];
  }
  als[t] = ss;
  ald[t] = sd;
}

__device__ __forceinline__ void edge_sd(const int* __restrict__ ei, int ne,
                                        int e, int& s, int& d) {
  if (e < ne) { s = ei[e]; d = ei[ne + e]; }
  else        { s = e - ne; d = s; }        // self-loops appended
}

__device__ __forceinline__ float lrelu(float x) {
  return x > 0.f ? x : NEG_SLOPE * x;
}

// pass 1: segment max over destination
__global__ void gat_edge_max(const int* __restrict__ ei, int ne, int nn,
                             const float* __restrict__ als,
                             const float* __restrict__ ald,
                             float* __restrict__ m, int Hn) {
  const long long t = (long long)blockIdx.x * blockDim.x + threadIdx.x;
  const long long total = (long long)(ne + nn) * Hn;
  if (t >= total) return;
  const int e = (int)(t / Hn), h = (int)(t - (long long)e * Hn);
  int s, d;
  edge_sd(ei, ne, e, s, d);
  const float el = lrelu(als[(size_t)s * Hn + h] + ald[(size_t)d * Hn + h]);
  __hip_atomic_fetch_max(&m[(size_t)d * Hn + h], el, __ATOMIC_RELAXED,
                         __HIP_MEMORY_SCOPE_AGENT);
}

// pass 2: p = exp(e - m[dst]); cache p per edge; segment-sum into ssum
__global__ void gat_edge_expsum(const int* __restrict__ ei, int ne, int nn,
                                const float* __restrict__ als,
                                const float* __restrict__ ald,
                                const float* __restrict__ m,
                                float* __restrict__ ssum,
                                float* __restrict__ pbuf, int Hn) {
  const long long t = (long long)blockIdx.x * blockDim.x + threadIdx.x;
  const long long total = (long long)(ne + nn) * Hn;
  if (t >= total) return;
  const int e = (int)(t / Hn), h = (int)(t - (long long)e * Hn);
  int s, d;
  edge_sd(ei, ne, e, s, d);
  const float el = lrelu(als[(size_t)s * Hn + h] + ald[(size_t)d * Hn + h]);
  const float p = expf(el - m[(size_t)d * Hn + h]);
  pbuf[t] = p;
  atomicAdd(&ssum[(size_t)d * Hn + h], p);
}

// pass 3: out[dst] += (p / (s[dst]+eps)) * h[src]   (one thread per edge-channel)
__global__ void gat_edge_msg(const int* __restrict__ ei, int ne, int nn,
                             const float* __restrict__ hsrc,
                             const float* __restrict__ pbuf,
                             const float* __restrict__ ssum,
                             float* __restrict__ acc, int Hn, int HIDn, int HCn) {
  const long long t = (long long)blockIdx.x * blockDim.x + threadIdx.x;
  const long long total = (long long)(ne + nn) * HCn;
  if (t >= total) return;
  const int e = (int)(t / HCn), c = (int)(t - (long long)e * HCn);
  const int h = c / HIDn;
  int s, d;
  edge_sd(ei, ne, e, s, d);
  const float p = pbuf[(size_t)e * Hn + h];
  const float alpha = p / (ssum[(size_t)d * Hn + h] + 1e-16f);
  atomicAdd(&acc[(size_t)d * HCn + c], alpha * hsrc[(size_t)s * HCn + c]);
}

// out = BN(acc + bias) -> ELU
__global__ void gat_bias_bn_elu(const float* __restrict__ acc,
                                const float* __restrict__ b,
                                const float* __restrict__ g,
                                const float* __restrict__ be,
                                const float* __restrict__ rm,
                                const float* __restrict__ rv,
                                float* __restrict__ o, int total, int HCn) {
  const int t = blockIdx.x * blockDim.x + threadIdx.x;
  if (t >= total) return;
  const int c = t % HCn;
  const float v = acc[t] + b[c];
  const float y = (v - rm[c]) * (g[c] * rsqrtf(rv[c] + BN_EPS)) + be[c];
  o[t] = y > 0.f ? y : ELU_ALPHA * expm1f(y);
}

__global__ void gat_add_bias(float* __restrict__ o, const float* __restrict__ b,
                             int total, int HCn) {
  const int t = blockIdx.x * blockDim.x + threadIdx.x;
  if (t >= total) return;
  o[t] += b[t % HCn];
}

// =====================================================================
static inline unsigned cdiv(long long a, long long b) {
  return (unsigned)((a + b - 1) / b);
}

extern "C" void kernel_launch(void* const* d_in, const int* in_sizes, int n_in,
                              void* d_out, int out_size, void* d_ws, size_t ws_size,
                              hipStream_t stream) {
  (void)in_sizes; (void)n_in; (void)out_size; (void)ws_size;
  const float* x   = (const float*)d_in[0];
  const int*   ei  = (const int*)d_in[1];
  const float* W0  = (const float*)d_in[2];
  const float* as0 = (const float*)d_in[3];
  const float* ad0 = (const float*)d_in[4];
  const float* b0  = (const float*)d_in[5];
  const float* g0  = (const float*)d_in[6];
  const float* be0 = (const float*)d_in[7];
  const float* rm0 = (const float*)d_in[8];
  const float* rv0 = (const float*)d_in[9];
  const float* W1  = (const float*)d_in[10];
  const float* as1 = (const float*)d_in[11];
  const float* ad1 = (const float*)d_in[12];
  const float* b1  = (const float*)d_in[13];
  const float* g1  = (const float*)d_in[14];
  const float* be1 = (const float*)d_in[15];
  const float* rm1 = (const float*)d_in[16];
  const float* rv1 = (const float*)d_in[17];
  const float* W2  = (const float*)d_in[18];
  const float* as2 = (const float*)d_in[19];
  const float* ad2 = (const float*)d_in[20];
  const float* b2  = (const float*)d_in[21];
  float* out = (float*)d_out;

  // workspace layout (floats)
  float* ws = (float*)d_ws;
  size_t off = 0;
  float* hbuf   = ws + off; off += (size_t)NNODES * HCC;   // post-linear features
  float* actbuf = ws + off; off += (size_t)NNODES * HCC;   // post BN+ELU activations
  float* accbuf = ws + off; off += (size_t)NNODES * HCC;   // scatter accumulator
  float* als    = ws + off; off += (size_t)NNODES * NHEAD;
  float* ald    = ws + off; off += (size_t)NNODES * NHEAD;
  float* mbuf   = ws + off; off += (size_t)NNODES * NHEAD;
  float* sbuf   = ws + off; off += (size_t)NNODES * NHEAD;
  float* pbuf   = ws + off; off += (size_t)(NEDGES + NNODES) * NHEAD;

  const long long Etot = (long long)NEDGES + NNODES;
  const int TB = 256;
  const float NEGINF = -3.402823466e38f;  // overwritten by self-loop for every node

  // ---------------- layer 0: F_IN -> (8,16) concat ----------------
  {
    dim3 g(cdiv(NNODES, 128), HCC / 16);
    gat_gemm_wmma<<<g, TB, F_INC * 16 * sizeof(float), stream>>>(x, W0, hbuf,
                                                                 NNODES, F_INC, HCC);
    gat_attn_logits<<<cdiv((long long)NNODES * NHEAD, TB), TB, 0, stream>>>(
        hbuf, as0, ad0, als, ald, NNODES, NHEAD, HIDC, HCC);
    fill_f32<<<cdiv(NNODES * NHEAD, TB), TB, 0, stream>>>(mbuf, NNODES * NHEAD, NEGINF);
    fill_f32<<<cdiv(NNODES * NHEAD, TB), TB, 0, stream>>>(sbuf, NNODES * NHEAD, 0.f);
    fill_f32<<<cdiv((long long)NNODES * HCC, TB), TB, 0, stream>>>(accbuf, NNODES * HCC, 0.f);
    gat_edge_max<<<cdiv(Etot * NHEAD, TB), TB, 0, stream>>>(ei, NEDGES, NNODES,
                                                            als, ald, mbuf, NHEAD);
    gat_edge_expsum<<<cdiv(Etot * NHEAD, TB), TB, 0, stream>>>(ei, NEDGES, NNODES,
                                                               als, ald, mbuf, sbuf,
                                                               pbuf, NHEAD);
    gat_edge_msg<<<cdiv(Etot * HCC, TB), TB, 0, stream>>>(ei, NEDGES, NNODES, hbuf,
                                                          pbuf, sbuf, accbuf,
                                                          NHEAD, HIDC, HCC);
    gat_bias_bn_elu<<<cdiv((long long)NNODES * HCC, TB), TB, 0, stream>>>(
        accbuf, b0, g0, be0, rm0, rv0, actbuf, NNODES * HCC, HCC);
  }

  // ---------------- layer 1: 128 -> (8,16) concat ----------------
  {
    dim3 g(cdiv(NNODES, 128), HCC / 16);
    gat_gemm_wmma<<<g, TB, HCC * 16 * sizeof(float), stream>>>(actbuf, W1, hbuf,
                                                               NNODES, HCC, HCC);
    gat_attn_logits<<<cdiv((long long)NNODES * NHEAD, TB), TB, 0, stream>>>(
        hbuf, as1, ad1, als, ald, NNODES, NHEAD, HIDC, HCC);
    fill_f32<<<cdiv(NNODES * NHEAD, TB), TB, 0, stream>>>(mbuf, NNODES * NHEAD, NEGINF);
    fill_f32<<<cdiv(NNODES * NHEAD, TB), TB, 0, stream>>>(sbuf, NNODES * NHEAD, 0.f);
    fill_f32<<<cdiv((long long)NNODES * HCC, TB), TB, 0, stream>>>(accbuf, NNODES * HCC, 0.f);
    gat_edge_max<<<cdiv(Etot * NHEAD, TB), TB, 0, stream>>>(ei, NEDGES, NNODES,
                                                            als, ald, mbuf, NHEAD);
    gat_edge_expsum<<<cdiv(Etot * NHEAD, TB), TB, 0, stream>>>(ei, NEDGES, NNODES,
                                                               als, ald, mbuf, sbuf,
                                                               pbuf, NHEAD);
    gat_edge_msg<<<cdiv(Etot * HCC, TB), TB, 0, stream>>>(ei, NEDGES, NNODES, hbuf,
                                                          pbuf, sbuf, accbuf,
                                                          NHEAD, HIDC, HCC);
    gat_bias_bn_elu<<<cdiv((long long)NNODES * HCC, TB), TB, 0, stream>>>(
        accbuf, b1, g1, be1, rm1, rv1, actbuf, NNODES * HCC, HCC);
  }

  // ---------------- layer 2: 128 -> (1,40), mean == identity ----------------
  {
    dim3 g(cdiv(NNODES, 128), (LBL + 15) / 16);
    gat_gemm_wmma<<<g, TB, HCC * 16 * sizeof(float), stream>>>(actbuf, W2, hbuf,
                                                               NNODES, HCC, LBL);
    gat_attn_logits<<<cdiv(NNODES, TB), TB, 0, stream>>>(hbuf, as2, ad2, als, ald,
                                                         NNODES, 1, LBL, LBL);
    fill_f32<<<cdiv(NNODES, TB), TB, 0, stream>>>(mbuf, NNODES, NEGINF);
    fill_f32<<<cdiv(NNODES, TB), TB, 0, stream>>>(sbuf, NNODES, 0.f);
    fill_f32<<<cdiv(NNODES * LBL, TB), TB, 0, stream>>>(out, NNODES * LBL, 0.f);
    gat_edge_max<<<cdiv(Etot, TB), TB, 0, stream>>>(ei, NEDGES, NNODES, als, ald,
                                                    mbuf, 1);
    gat_edge_expsum<<<cdiv(Etot, TB), TB, 0, stream>>>(ei, NEDGES, NNODES, als, ald,
                                                       mbuf, sbuf, pbuf, 1);
    gat_edge_msg<<<cdiv(Etot * LBL, TB), TB, 0, stream>>>(ei, NEDGES, NNODES, hbuf,
                                                          pbuf, sbuf, out, 1, LBL, LBL);
    gat_add_bias<<<cdiv(NNODES * LBL, TB), TB, 0, stream>>>(out, b2, NNODES * LBL, LBL);
  }
}
